// DGCNN_sub_old_7842610283367
// MI455X (gfx1250) — matrix operand order repaired and checked by reading, not compiled, not run on hardware
//
#include <hip/hip_runtime.h>
#include <math.h>

typedef __attribute__((ext_vector_type(16))) __bf16 v16bf;
typedef __attribute__((ext_vector_type(8)))  __bf16 v8bf;
typedef __attribute__((ext_vector_type(8)))  float  v8f;

#define NNODES 102400
#define DCAT   97
#define NSUB   2048   // B*S = 64*32
#define MSZ    50
#define KPOOL  30

static __device__ __forceinline__ __bf16 f2bf(float f) {
  union { float f; unsigned u; } a; a.f = f;
  unsigned r = a.u + 0x7fffu + ((a.u >> 16) & 1u);   // round-to-nearest-even
  unsigned short h = (unsigned short)(r >> 16);
  __bf16 o; __builtin_memcpy(&o, &h, 2); return o;
}

__global__ void k_zero_i32(int* __restrict__ p, int n) {
  int i = blockIdx.x * blockDim.x + threadIdx.x;
  if (i < n) p[i] = 0;
}

__global__ void k_count(const int* __restrict__ col, int* __restrict__ degi, long long E) {
  long long e = (long long)blockIdx.x * blockDim.x + threadIdx.x;
  if (e < E) atomicAdd(&degi[col[e]], 1);
}

__global__ void k_dinv(const int* __restrict__ degi, float* __restrict__ dinv, int n) {
  int i = blockIdx.x * blockDim.x + threadIdx.x;
  if (i < n) dinv[i] = rsqrtf(1.0f + (float)degi[i]);   // self-loop guarantees deg>=1
}

// dst (compact bf16 [rows,w]) = src f32 [rows, lds-strided], taking first w cols
__global__ void k_cvt_bf16(const float* __restrict__ src, int lds,
                           __bf16* __restrict__ dst, int w, long long total) {
  long long i = (long long)blockIdx.x * blockDim.x + threadIdx.x;
  if (i >= total) return;
  long long r = i / w; int c = (int)(i % w);
  dst[i] = f2bf(src[r * lds + c]);
}

// Bt[n*Kd + k] = bf16(B[k*Nd + n])  (transpose to column-major for B fragments)
__global__ void k_cvtT_bf16(const float* __restrict__ B, int Kd, int Nd,
                            __bf16* __restrict__ Bt) {
  int i = blockIdx.x * blockDim.x + threadIdx.x;
  if (i >= Kd * Nd) return;
  int n = i / Kd, k = i % Kd;
  Bt[i] = f2bf(B[(long long)k * Nd + n]);
}

// C[M,N] = A[M,K] @ B[K,N]; bf16 inputs (A row-major, B pre-transposed col-major),
// f32 accumulate via v_wmma_f32_16x16x32_bf16. One wave computes a 16 x (16*NT)
// C tile (A fragment reused across NT WMMAs). M%16==0, N%(16*NT)==0, K%32==0,
// lda/ldbt multiples of 8 (16B-aligned vector loads).
// Fragment layouts used here (wave32; A and B share contiguous k order):
//   A (16x32): elem e -> K = half*16 + e, M = lane&15
//              => two contiguous 8-elem (16B) loads per K-step
//   B (32x16): elem e -> K = half*16 + e, N = lane&15
//              => one contiguous 16-elem (32B) load per K-step from Bt
//   C (16x16): vgpr v -> M = v + half*8, N = lane&15
template <int NT>
__global__ void k_gemm_bf16(const __bf16* __restrict__ A, int lda,
                            const __bf16* __restrict__ Bt, int ldbt,
                            float* __restrict__ C, int ldc,
                            int mtiles, int ngroups, int K) {
  int wave = (int)((blockIdx.x * blockDim.x + threadIdx.x) >> 5);
  int lane = threadIdx.x & 31;
  int tm = wave / ngroups, tn0 = (wave % ngroups) * NT;
  if (tm >= mtiles) return;                 // wave-uniform: EXEC stays all-ones
  int half = lane >> 4, r = lane & 15;
  const __bf16* ap = A + (long long)(tm * 16 + r) * lda + half * 16;
  const __bf16* bp = Bt + (long long)(tn0 * 16 + r) * ldbt + half * 16;
  v8f acc[NT];
#pragma unroll
  for (int nt = 0; nt < NT; ++nt) acc[nt] = (v8f){};
#pragma unroll 2
  for (int k0 = 0; k0 < K; k0 += 32) {
    v8bf alo = *(const v8bf*)(ap + k0);
    v8bf ahi = *(const v8bf*)(ap + k0 + 8);
    v16bf a = __builtin_shufflevector(alo, ahi,
                                      0, 1, 2, 3, 4, 5, 6, 7,
                                      8, 9, 10, 11, 12, 13, 14, 15);
#pragma unroll
    for (int nt = 0; nt < NT; ++nt) {
      v16bf b = *(const v16bf*)(bp + (long long)nt * 16 * ldbt + k0);
      acc[nt] = __builtin_amdgcn_wmma_f32_16x16x32_bf16(false, a, false, b,
                                                        (short)0, acc[nt],
                                                        false, false);
    }
  }
#pragma unroll
  for (int nt = 0; nt < NT; ++nt)
#pragma unroll
    for (int v = 0; v < 8; ++v)
      C[(long long)(tm * 16 + v + half * 8) * ldc + (tn0 + nt) * 16 + r] = acc[nt][v];
}

// out[n] = dot(A[n, 0:32], w[0:32])  (layer-4 GEMV, dout == 1)
__global__ void k_gemv32(const float* __restrict__ A, int lda,
                         const float* __restrict__ w, float* __restrict__ out, int n) {
  int i = blockIdx.x * blockDim.x + threadIdx.x;
  if (i >= n) return;
  const float* ar = A + (long long)i * lda;
  float s = 0.f;
#pragma unroll
  for (int k = 0; k < 32; ++k) s += ar[k] * w[k];
  out[i] = s;
}

// agg[n,d] = dinv[n]^2 * XW[n,d]   (self-loop contribution, initializes agg)
template <int DOUT>
__global__ void k_selfinit(const float* __restrict__ XW, const float* __restrict__ dinv,
                           float* __restrict__ agg, int n) {
  long long i = (long long)blockIdx.x * blockDim.x + threadIdx.x;
  if (i >= (long long)n * DOUT) return;
  int node = (int)(i / DOUT);
  float di = dinv[node];
  agg[i] = di * di * XW[i];
}

// agg[col,:] += dinv[row]*dinv[col] * XW[row,:]
// wave-uniform edge id -> row/col/dinv become scalar loads; lane == feature dim
__global__ void k_scatter32(const float* __restrict__ XW, const int* __restrict__ row,
                            const int* __restrict__ col, const float* __restrict__ dinv,
                            float* __restrict__ agg, long long E) {
  long long e = (long long)blockIdx.x * (blockDim.x >> 5) + (threadIdx.x >> 5);
  int d = threadIdx.x & 31;
  if (e >= E) return;                       // wave-uniform guard
  int rr = row[e], cc = col[e];
  float nrm = dinv[rr] * dinv[cc];
  atomicAdd(&agg[(long long)cc * 32 + d], nrm * XW[(long long)rr * 32 + d]);
}

__global__ void k_scatter1(const float* __restrict__ XW, const int* __restrict__ row,
                           const int* __restrict__ col, const float* __restrict__ dinv,
                           float* __restrict__ agg, long long E) {
  long long e = (long long)blockIdx.x * blockDim.x + threadIdx.x;
  if (e >= E) return;
  int rr = row[e], cc = col[e];
  atomicAdd(&agg[cc], dinv[rr] * dinv[cc] * XW[rr]);
}

// cs[n, coff+d] = tanh(agg[n,d] + bias[d])
template <int DOUT>
__global__ void k_finalize(const float* __restrict__ agg, const float* __restrict__ bias,
                           float* __restrict__ cs, int coff, int n) {
  long long i = (long long)blockIdx.x * blockDim.x + threadIdx.x;
  if (i >= (long long)n * DOUT) return;
  int node = (int)(i / DOUT);
  int d = (int)(i % DOUT);
  cs[(long long)node * DCAT + coff + d] = tanhf(agg[i] + bias[d]);
}

// Per (graph,sub) block: sort-pool top-30, conv1(1->16,k=97)+relu, maxpool2,
// conv2(16->32,k=5)+relu, write flattened [352] (channel-major).
__global__ void k_sortpool(const float* __restrict__ cs,
                           const float* __restrict__ c1w, const float* __restrict__ c1b,
                           const float* __restrict__ c2w, const float* __restrict__ c2b,
                           float* __restrict__ flat) {
  int g = blockIdx.x;
  int t = threadIdx.x;
  __shared__ float val[MSZ];
  __shared__ int   slot[KPOOL];
  __shared__ float h1[16][KPOOL];
  __shared__ float pool[16][15];
  const float* base = cs + (long long)g * MSZ * DCAT;
  if (t < MSZ) val[t] = base[t * DCAT + (DCAT - 1)];
  __syncthreads();
  if (t < MSZ) {
    float v = val[t];
    int rank = 0;
    for (int j = 0; j < MSZ; ++j) {
      float vj = val[j];
      if (vj > v || (vj == v && j < t)) rank++;   // stable desc == argsort(-x)
    }
    if (rank < KPOOL) slot[rank] = t;
  }
  __syncthreads();
  for (int i = t; i < 16 * KPOOL; i += blockDim.x) {
    int o = i / KPOOL, k = i % KPOOL;
    const float* nrow = base + (long long)slot[k] * DCAT;
    const float* wrow = c1w + o * DCAT;
    float s = c1b[o];
    for (int d = 0; d < DCAT; ++d) s += nrow[d] * wrow[d];
    h1[o][k] = fmaxf(s, 0.0f);
  }
  __syncthreads();
  for (int i = t; i < 16 * 15; i += blockDim.x) {
    int o = i / 15, j = i % 15;
    pool[o][j] = fmaxf(h1[o][2 * j], h1[o][2 * j + 1]);
  }
  __syncthreads();
  for (int i = t; i < 352; i += blockDim.x) {
    int c2 = i / 11, pos = i % 11;
    float s = c2b[c2];
    for (int c1 = 0; c1 < 16; ++c1) {
      const float* w = c2w + (c2 * 16 + c1) * 5;
#pragma unroll
      for (int kk = 0; kk < 5; ++kk) s += pool[c1][pos + kk] * w[kk];
    }
    flat[(long long)g * 352 + i] = fmaxf(s, 0.0f);
  }
}

// Per-graph: mean over 32 subgraphs (+lin1 bias), relu, lin2, log_softmax.
__global__ void k_head(const float* __restrict__ h3, const float* __restrict__ l1b,
                       const float* __restrict__ l2w, const float* __restrict__ l2b,
                       float* __restrict__ out) {
  int gph = blockIdx.x;
  int t = threadIdx.x;
  __shared__ float acc[128];
  __shared__ float logits[10];
  float s = 0.f;
  for (int sb = 0; sb < 32; ++sb)
    s += h3[((long long)gph * 32 + sb) * 128 + t];
  s = s * (1.0f / 32.0f) + l1b[t];
  acc[t] = fmaxf(s, 0.0f);
  __syncthreads();
  if (t < 10) {
    float z = l2b[t];
    for (int d = 0; d < 128; ++d) z += acc[d] * l2w[d * 10 + t];
    logits[t] = z;
  }
  __syncthreads();
  if (t == 0) {
    float m = logits[0];
    for (int i = 1; i < 10; ++i) m = fmaxf(m, logits[i]);
    float se = 0.f;
    for (int i = 0; i < 10; ++i) se += expf(logits[i] - m);
    float lse = m + logf(se);
    for (int i = 0; i < 10; ++i) out[gph * 10 + i] = logits[i] - lse;
  }
}

extern "C" void kernel_launch(void* const* d_in, const int* in_sizes, int n_in,
                              void* d_out, int out_size, void* d_ws, size_t ws_size,
                              hipStream_t stream) {
  (void)n_in; (void)out_size; (void)ws_size;
  const float* x    = (const float*)d_in[0];          // [102400, 256]
  const int*   ei   = (const int*)d_in[1];            // [2, E]
  const long long E = (long long)(in_sizes[1] / 2);
  const int* row = ei;
  const int* col = ei + E;
  const float* W[4] = { (const float*)d_in[2], (const float*)d_in[4],
                        (const float*)d_in[6], (const float*)d_in[8] };
  const float* Bv[4] = { (const float*)d_in[3], (const float*)d_in[5],
                         (const float*)d_in[7], (const float*)d_in[9] };
  const float* c1w = (const float*)d_in[10];
  const float* c1b = (const float*)d_in[11];
  const float* c2w = (const float*)d_in[12];
  const float* c2b = (const float*)d_in[13];
  const float* l1w = (const float*)d_in[14];
  const float* l1b = (const float*)d_in[15];
  const float* l2w = (const float*)d_in[16];
  const float* l2b = (const float*)d_in[17];
  float* out = (float*)d_out;

  const int N = NNODES;
  char* ws = (char*)d_ws;
  int*    degi = (int*)(ws + 0);                      //  0.41 MB
  float*  dinv = (float*)(ws + 409600);               //  0.41 MB
  float*  cs   = (float*)(ws + 819200);               // 39.73 MB  [N,97]
  float*  XW   = (float*)(ws + 40550400);             // 13.11 MB  [N,32]
  float*  agg  = (float*)(ws + 53657600);             // 13.11 MB  [N,32]
  __bf16* Abf  = (__bf16*)(ws + 66764800);            // 52.43 MB  (bf16 A staging)
  __bf16* Btbf = (__bf16*)(ws + 119193600);           //  0.09 MB  (bf16 B^T staging)
  float* flat = XW;                                   // reuse: [2048,352]
  float* h3   = agg;                                  // reuse: [2048,128]

  const int TB = 256;
  // --- degree / dinv ---
  k_zero_i32<<<(N + TB - 1) / TB, TB, 0, stream>>>(degi, N);
  k_count<<<(unsigned)((E + TB - 1) / TB), TB, 0, stream>>>(col, degi, E);
  k_dinv<<<(N + TB - 1) / TB, TB, 0, stream>>>(degi, dinv, N);

  // --- GCN layers 0..2 (dout=32): bf16 staging + WMMA GEMM, then aggregate ---
  const int mtiles = N / 16;                          // 6400
  const unsigned gemm_blocks = (unsigned)((long long)mtiles * 32 / TB);  // NT=2, ngroups=1
  const unsigned nd_blocks   = (unsigned)(((long long)N * 32 + TB - 1) / TB);
  const unsigned sc_blocks   = (unsigned)((E + (TB / 32) - 1) / (TB / 32));
  for (int l = 0; l < 3; ++l) {
    const float* Af  = (l == 0) ? x : (cs + (l - 1) * 32);
    int          lds = (l == 0) ? 256 : DCAT;
    int          K   = (l == 0) ? 256 : 32;
    long long tot = (long long)N * K;
    k_cvt_bf16<<<(unsigned)((tot + TB - 1) / TB), TB, 0, stream>>>(Af, lds, Abf, K, tot);
    k_cvtT_bf16<<<(K * 32 + TB - 1) / TB, TB, 0, stream>>>(W[l], K, 32, Btbf);
    k_gemm_bf16<2><<<gemm_blocks, TB, 0, stream>>>(Abf, K, Btbf, K, XW, 32,
                                                   mtiles, /*ngroups=*/1, K);
    k_selfinit<32><<<nd_blocks, TB, 0, stream>>>(XW, dinv, agg, N);
    k_scatter32<<<sc_blocks, TB, 0, stream>>>(XW, row, col, dinv, agg, E);
    k_finalize<32><<<nd_blocks, TB, 0, stream>>>(agg, Bv[l], cs, l * 32, N);
  }
  // --- GCN layer 3 (dout=1) ---
  k_gemv32<<<(N + TB - 1) / TB, TB, 0, stream>>>(cs + 64, DCAT, W[3], XW, N);
  k_selfinit<1><<<(N + TB - 1) / TB, TB, 0, stream>>>(XW, dinv, agg, N);
  k_scatter1<<<(unsigned)((E + TB - 1) / TB), TB, 0, stream>>>(XW, row, col, dinv, agg, E);
  k_finalize<1><<<(N + TB - 1) / TB, TB, 0, stream>>>(agg, Bv[3], cs, 96, N);

  // --- sort-pool + conv1 + maxpool + conv2 -> flat [2048,352] ---
  k_sortpool<<<NSUB, 128, 0, stream>>>(cs, c1w, c1b, c2w, c2b, flat);

  // --- lin1: [2048,352] @ [352,128] via WMMA (NT=2 -> 4 tile-groups wide) ---
  {
    long long tot = (long long)NSUB * 352;
    k_cvt_bf16<<<(unsigned)((tot + TB - 1) / TB), TB, 0, stream>>>(flat, 352, Abf, 352, tot);
    k_cvtT_bf16<<<(352 * 128 + TB - 1) / TB, TB, 0, stream>>>(l1w, 352, 128, Btbf);
    const unsigned blocks = (unsigned)((long long)(NSUB / 16) * 4 * 32 / TB);
    k_gemm_bf16<2><<<blocks, TB, 0, stream>>>(Abf, 352, Btbf, 352, h3, 128,
                                              NSUB / 16, /*ngroups=*/4, 352);
  }

  // --- head: mean + relu + lin2 + log_softmax ---
  k_head<<<64, 128, 0, stream>>>(h3, l1b, l2w, l2b, out);
}
